// DiffPool_85959475462402
// MI455X (gfx1250) — matrix operand order repaired and checked by reading, not compiled, not run on hardware
//
#include <hip/hip_runtime.h>
#include <hip/hip_bf16.h>

// ---------------- problem constants (match reference) ----------------
#define NATOMS   16384      // B * N_PER_MOL
#define NPAIRS   131072
#define D_ATOM   256
#define D_PAIR   64
#define NMOL     256
#define NPER     64
#define NCLUST   128
#define ACAT_LD  160        // 100 (AA) + 50 (PA) + 10 pad
#define APP_LD   128        // 50 (AP) + 50 (PP) + 28 pad
#define POOL_ELEMS ((size_t)NMOL * NCLUST * D_ATOM)   // 8388608

typedef __attribute__((ext_vector_type(16))) __bf16 v16bf;
typedef __attribute__((ext_vector_type(8)))  float  v8f;
typedef __attribute__((ext_vector_type(4)))  float  f32x4;

// Build bf16 A-fragment (contiguous k order, same order as the packed B tiles) from a row-major f32 row.
// lane<16: e0..15 <- K=k0+0..15
// lane>=16: e0..15 <- K=k0+16..31
__device__ __forceinline__ v16bf load_a_frag(const float* __restrict__ rowp, int k0, int half) {
    const f32x4* p1 = (const f32x4*)(rowp + k0 + half * 16);
    const f32x4* p2 = (const f32x4*)(rowp + k0 + 8 + half * 16);
    f32x4 a0 = p1[0], a1 = p1[1], a2 = p2[0], a3 = p2[1];
    v16bf a;
    a[0]  = (__bf16)a0[0]; a[1]  = (__bf16)a0[1]; a[2]  = (__bf16)a0[2]; a[3]  = (__bf16)a0[3];
    a[4]  = (__bf16)a1[0]; a[5]  = (__bf16)a1[1]; a[6]  = (__bf16)a1[2]; a[7]  = (__bf16)a1[3];
    a[8]  = (__bf16)a2[0]; a[9]  = (__bf16)a2[1]; a[10] = (__bf16)a2[2]; a[11] = (__bf16)a2[3];
    a[12] = (__bf16)a3[0]; a[13] = (__bf16)a3[1]; a[14] = (__bf16)a3[2]; a[15] = (__bf16)a3[3];
    return a;
}

// ---------------- workspace zero (ACAT only: atomic segment-sum target) --------
__global__ void zero_kernel(float* __restrict__ p, size_t n) {
    size_t i = (size_t)blockIdx.x * blockDim.x + threadIdx.x;
    size_t st = (size_t)gridDim.x * blockDim.x;
    for (; i < n; i += st) p[i] = 0.f;
}

// ---------------- weight pre-pack: f32 [Kw x Nw] -> bf16 fragment-order tiles ----
// Tile (kt,nt) holds the wave32 B-fragment in contiguous k order: lane<16: n=lane, K=kt*32+e;
// lane>=16: n=lane-16, K=kt*32+16+e. Out-of-range -> 0 (kills all hot-loop masks).
__global__ void pack_w_kernel(const float* __restrict__ W, int Kw, int Nw,
                              __bf16* __restrict__ dst, int KT)
{
    const int kt = blockIdx.x, nt = blockIdx.y;
    const int lane = threadIdx.x;
    const int n  = nt * 16 + (lane & 15);
    const int kb = kt * 32 + (lane >> 4) * 16;
    v16bf f;
#pragma unroll
    for (int e = 0; e < 16; ++e) {
        int k = kb + e;
        f[e] = (k < Kw && n < Nw) ? (__bf16)W[(size_t)k * Nw + n] : (__bf16)0.f;
    }
    *(v16bf*)(dst + ((size_t)nt * KT + kt) * 512 + lane * 16) = f;
}

__global__ void pad_bias_kernel(const float* __restrict__ b, int n,
                                float* __restrict__ dst, int npad)
{
    int t = threadIdx.x;
    if (t < npad) dst[t] = (t < n) ? b[t] : 0.f;
}

// ---------------- generic GEMM + bias + relu (no LDS, no bounds) ----------------
// out[row, col_off + n0+nl] = relu(A[row,:] @ Wpacked[:, col] + bias[col])
template<int KT>
__global__ void gemm_bf16_kernel(const float* __restrict__ A, int lda,
                                 const __bf16* __restrict__ Wp,
                                 const float* __restrict__ biasp,
                                 float* __restrict__ out, int ldo, int col_off)
{
    const int wave = threadIdx.x >> 5, lane = threadIdx.x & 31;
    const int half = lane >> 4, nl = lane & 15;
    const int row0 = (blockIdx.x * 8 + wave) * 16;
    const int n0   = blockIdx.y * 16;
    const float* arow = A + (size_t)(row0 + nl) * lda;
    const __bf16* wp  = Wp + ((size_t)blockIdx.y * KT) * 512 + lane * 16;
    v8f acc = {};
#pragma unroll
    for (int kt = 0; kt < KT; ++kt) {
        v16bf b = *(const v16bf*)(wp + kt * 512);
        v16bf a = load_a_frag(arow, kt * 32, half);
        acc = __builtin_amdgcn_wmma_f32_16x16x32_bf16(false, a, false, b, (short)0, acc, false, false);
    }
    const int col = n0 + nl;
    const float bia = biasp[col];
    float* op = out + (size_t)(row0 + half * 8) * ldo + col_off + col;
#pragma unroll
    for (int r = 0; r < 8; ++r)
        op[(size_t)r * ldo] = fmaxf(acc[r] + bia, 0.f);
}

// ---------------- PA GEMM fused with segment-sum scatter (KT=2) ----------------
__global__ void gemm_pa_scatter_kernel(const float* __restrict__ pair,
                                       const __bf16* __restrict__ Wp,
                                       const float* __restrict__ biasp,
                                       const int* __restrict__ pair_split,
                                       float* __restrict__ ACAT)
{
    const int wave = threadIdx.x >> 5, lane = threadIdx.x & 31;
    const int half = lane >> 4, nl = lane & 15;
    const int row0 = (blockIdx.x * 8 + wave) * 16;
    const int n0   = blockIdx.y * 16;
    const float* arow = pair + (size_t)(row0 + nl) * D_PAIR;
    const __bf16* wp  = Wp + ((size_t)blockIdx.y * 2) * 512 + lane * 16;
    v8f acc = {};
#pragma unroll
    for (int kt = 0; kt < 2; ++kt) {
        v16bf b = *(const v16bf*)(wp + kt * 512);
        v16bf a = load_a_frag(arow, kt * 32, half);
        acc = __builtin_amdgcn_wmma_f32_16x16x32_bf16(false, a, false, b, (short)0, acc, false, false);
    }
    const int col = n0 + nl;
    if (col < 50) {
        const float bia = biasp[col];
#pragma unroll
        for (int r = 0; r < 8; ++r) {
            int row = row0 + half * 8 + r;
            float v = fmaxf(acc[r] + bia, 0.f);
            atomicAdd(&ACAT[(size_t)pair_split[row] * ACAT_LD + 100 + col], v);
        }
    }
}

// ---------------- AP GEMM: dual gathered A-operands share every B fragment ------
// AP[p] = relu([a_i|a_j]W + b) + relu([a_j|a_i]W + b). K split at 256 -> no branch.
__global__ void gemm_ap_kernel(const float* __restrict__ atom,
                               const int* __restrict__ a2p,
                               const __bf16* __restrict__ Wp,
                               const float* __restrict__ biasp,
                               float* __restrict__ APP)
{
    const int wave = threadIdx.x >> 5, lane = threadIdx.x & 31;
    const int half = lane >> 4, nl = lane & 15;
    const int row0 = (blockIdx.x * 8 + wave) * 16;
    const int n0   = blockIdx.y * 16;
    const int p    = row0 + nl;
    const int i0 = a2p[2 * p], i1 = a2p[2 * p + 1];
    const float* pi = atom + (size_t)i0 * D_ATOM;
    const float* pj = atom + (size_t)i1 * D_ATOM;
    const __bf16* wp = Wp + ((size_t)blockIdx.y * 16) * 512 + lane * 16;
    v8f acc1 = {}, acc2 = {};
#pragma unroll
    for (int kt = 0; kt < 8; ++kt) {          // K = 0..255: ij -> a_i, ji -> a_j
        v16bf b  = *(const v16bf*)(wp + kt * 512);
        v16bf a1 = load_a_frag(pi, kt * 32, half);
        v16bf a2 = load_a_frag(pj, kt * 32, half);
        acc1 = __builtin_amdgcn_wmma_f32_16x16x32_bf16(false, a1, false, b, (short)0, acc1, false, false);
        acc2 = __builtin_amdgcn_wmma_f32_16x16x32_bf16(false, a2, false, b, (short)0, acc2, false, false);
    }
#pragma unroll
    for (int kt = 8; kt < 16; ++kt) {         // K = 256..511: ij -> a_j, ji -> a_i
        v16bf b  = *(const v16bf*)(wp + kt * 512);
        v16bf a1 = load_a_frag(pj, (kt - 8) * 32, half);
        v16bf a2 = load_a_frag(pi, (kt - 8) * 32, half);
        acc1 = __builtin_amdgcn_wmma_f32_16x16x32_bf16(false, a1, false, b, (short)0, acc1, false, false);
        acc2 = __builtin_amdgcn_wmma_f32_16x16x32_bf16(false, a2, false, b, (short)0, acc2, false, false);
    }
    const int col = n0 + nl;
    if (col < 50) {
        const float bia = biasp[col];
#pragma unroll
        for (int r = 0; r < 8; ++r) {
            int row = row0 + half * 8 + r;
            APP[(size_t)row * APP_LD + col] = fmaxf(acc1[r] + bia, 0.f) + fmaxf(acc2[r] + bia, 0.f);
        }
    }
}

// ---------------- row softmax over [N,128], one wave per row ----------------
__global__ void softmax_kernel(float* __restrict__ A, int Nrows) {
    int row  = blockIdx.x * (blockDim.x >> 5) + (threadIdx.x >> 5);
    int lane = threadIdx.x & 31;
    if (row >= Nrows) return;
    float* p = A + (size_t)row * NCLUST;
    float x0 = p[lane], x1 = p[lane + 32], x2 = p[lane + 64], x3 = p[lane + 96];
    float m = fmaxf(fmaxf(x0, x1), fmaxf(x2, x3));
    for (int o = 16; o > 0; o >>= 1) m = fmaxf(m, __shfl_xor(m, o, 32));
    float e0 = __expf(x0 - m), e1 = __expf(x1 - m), e2 = __expf(x2 - m), e3 = __expf(x3 - m);
    float s = e0 + e1 + e2 + e3;
    for (int o = 16; o > 0; o >>= 1) s += __shfl_xor(s, o, 32);
    float inv = 1.f / s;
    p[lane] = e0 * inv; p[lane + 32] = e1 * inv; p[lane + 64] = e2 * inv; p[lane + 96] = e3 * inv;
}

// ---------------- DiffPool: pooled[b] = S[b]^T @ X[b] (transpose via LDS) -------
__global__ void pool_kernel(const float* __restrict__ assign,
                            const float* __restrict__ atom,
                            float* __restrict__ pooled)
{
    __shared__ float sA[512];   // [m][k] = S[k][m]
    __shared__ float sB[512];   // [k][n] = X[k][n]
    const int lane = threadIdx.x & 31;
    const int half = lane >> 4, nl = lane & 15;
    const int mol = blockIdx.x;
    const int m0  = blockIdx.y * 16;
    const int n0  = blockIdx.z * 16;
    v8f acc = {};
    for (int k0 = 0; k0 < NPER; k0 += 32) {
        for (int i = lane; i < 512; i += 32) {
            int m = i >> 5, k = i & 31;
            sA[i] = assign[(size_t)(mol * NPER + k0 + k) * NCLUST + m0 + m];
        }
        for (int i = lane; i < 512; i += 32) {
            int k = i >> 4, n = i & 15;
            sB[i] = atom[(size_t)(mol * NPER + k0 + k) * D_ATOM + n0 + n];
        }
        __syncthreads();
        v16bf a, b;
#pragma unroll
        for (int e = 0; e < 16; ++e) {
            int ka = (half << 4) + e;
            a[e] = (__bf16)sA[nl * 32 + ka];
            int kb = (half << 4) + e;
            b[e] = (__bf16)sB[kb * 16 + nl];
        }
        acc = __builtin_amdgcn_wmma_f32_16x16x32_bf16(false, a, false, b, (short)0, acc, false, false);
        __syncthreads();
    }
#pragma unroll
    for (int r = 0; r < 8; ++r) {
        int m = m0 + half * 8 + r;
        pooled[((size_t)mol * NCLUST + m) * D_ATOM + n0 + nl] = acc[r];
    }
}

extern "C" void kernel_launch(void* const* d_in, const int* in_sizes, int n_in,
                              void* d_out, int out_size, void* d_ws, size_t ws_size,
                              hipStream_t stream) {
    (void)in_sizes; (void)n_in; (void)out_size; (void)ws_size;
    const float* atom       = (const float*)d_in[0];
    const float* pair       = (const float*)d_in[1];
    const int*   pair_split = (const int*)  d_in[2];
    const int*   a2p        = (const int*)  d_in[3];
    const float* W_aa = (const float*)d_in[5];  const float* b_aa = (const float*)d_in[6];
    const float* W_pa = (const float*)d_in[7];  const float* b_pa = (const float*)d_in[8];
    const float* W_ao = (const float*)d_in[9];  const float* b_ao = (const float*)d_in[10];
    const float* W_ap = (const float*)d_in[11]; const float* b_ap = (const float*)d_in[12];
    const float* W_pp = (const float*)d_in[13]; const float* b_pp = (const float*)d_in[14];
    const float* W_po = (const float*)d_in[15]; const float* b_po = (const float*)d_in[16];

    float* out    = (float*)d_out;
    float* pout   = out + POOL_ELEMS;
    float* ACAT   = (float*)d_ws;                              // [N,160] f32 (zeroed: atomics)
    float* APP    = ACAT + (size_t)NATOMS * ACAT_LD;           // [P,128] f32 (garbage pads x zero-W -> 0)
    float* ASSIGN = APP  + (size_t)NPAIRS * APP_LD;            // [N,128] f32
    __bf16* WPK   = (__bf16*)(ASSIGN + (size_t)NATOMS * NCLUST);
    __bf16* Waa_p = WPK;            // KT=8,  NT=7
    __bf16* Wpa_p = WPK + 28672;    // KT=2,  NT=4
    __bf16* Wao_p = WPK + 32768;    // KT=5,  NT=8
    __bf16* Wap_p = WPK + 53248;    // KT=16, NT=4
    __bf16* Wpp_p = WPK + 86016;    // KT=2,  NT=4
    __bf16* Wpo_p = WPK + 90112;    // KT=4,  NT=4
    float* BIASP  = (float*)(WPK + 98304);
    float* baa_p = BIASP;        float* bpa_p = BIASP + 112;
    float* bao_p = BIASP + 176;  float* bap_p = BIASP + 304;
    float* bpp_p = BIASP + 368;  float* bpo_p = BIASP + 432;

    // --- prep: zero segment-sum target; pack weights/biases (tiny, one-shot) ---
    zero_kernel<<<512, 256, 0, stream>>>(ACAT, (size_t)NATOMS * ACAT_LD);
    pack_w_kernel<<<dim3(8, 7),  32, 0, stream>>>(W_aa, 256, 100, Waa_p, 8);
    pack_w_kernel<<<dim3(2, 4),  32, 0, stream>>>(W_pa,  64,  50, Wpa_p, 2);
    pack_w_kernel<<<dim3(5, 8),  32, 0, stream>>>(W_ao, 150, 128, Wao_p, 5);
    pack_w_kernel<<<dim3(16, 4), 32, 0, stream>>>(W_ap, 512,  50, Wap_p, 16);
    pack_w_kernel<<<dim3(2, 4),  32, 0, stream>>>(W_pp,  64,  50, Wpp_p, 2);
    pack_w_kernel<<<dim3(4, 4),  32, 0, stream>>>(W_po, 100,  64, Wpo_p, 4);
    pad_bias_kernel<<<1, 128, 0, stream>>>(b_aa, 100, baa_p, 112);
    pad_bias_kernel<<<1, 128, 0, stream>>>(b_pa,  50, bpa_p,  64);
    pad_bias_kernel<<<1, 128, 0, stream>>>(b_ao, 128, bao_p, 128);
    pad_bias_kernel<<<1, 128, 0, stream>>>(b_ap,  50, bap_p,  64);
    pad_bias_kernel<<<1, 128, 0, stream>>>(b_pp,  50, bpp_p,  64);
    pad_bias_kernel<<<1, 128, 0, stream>>>(b_po,  64, bpo_p,  64);

    // --- main pipeline ---
    // AA = relu(atom @ W_aa + b) -> ACAT[:,0:100] (pad cols write exact zeros, pre-scatter)
    gemm_bf16_kernel<8><<<dim3(NATOMS/128, 7), 256, 0, stream>>>(atom, D_ATOM, Waa_p, baa_p, ACAT, ACAT_LD, 0);
    // PA segment-sum -> ACAT[:,100:150]
    gemm_pa_scatter_kernel<<<dim3(NPAIRS/128, 4), 256, 0, stream>>>(pair, Wpa_p, bpa_p, pair_split, ACAT);
    // A = relu(ACAT[:,0:160] @ W_ao + b) -> ASSIGN (W rows >=150 packed as zero)
    gemm_bf16_kernel<5><<<dim3(NATOMS/128, 8), 256, 0, stream>>>(ACAT, ACAT_LD, Wao_p, bao_p, ASSIGN, NCLUST, 0);
    softmax_kernel<<<NATOMS/4, 128, 0, stream>>>(ASSIGN, NATOMS);
    // AP dual-gather GEMM -> APP[:,0:50]
    gemm_ap_kernel<<<dim3(NPAIRS/128, 4), 256, 0, stream>>>(atom, a2p, Wap_p, bap_p, APP);
    // PP -> APP[:,50:100] (tiles also write zeros to 100..113)
    gemm_bf16_kernel<2><<<dim3(NPAIRS/128, 4), 256, 0, stream>>>(pair, D_PAIR, Wpp_p, bpp_p, APP, APP_LD, 50);
    // P_out = relu(APP[:,0:128] @ W_po + b) -> d_out tail (W rows >=100 zero kill pad garbage)
    gemm_bf16_kernel<4><<<dim3(NPAIRS/128, 4), 256, 0, stream>>>(APP, APP_LD, Wpo_p, bpo_p, pout, D_PAIR, 0);
    // pooled[b] = S^T X -> d_out head
    pool_kernel<<<dim3(NMOL, NCLUST/16, D_ATOM/16), 32, 0, stream>>>(ASSIGN, atom, out);
}
